// GNNLayer_6373731467382
// MI455X (gfx1250) — matrix operand order repaired and checked by reading, not compiled, not run on hardware
//
#include <hip/hip_runtime.h>

// ---------------- problem constants ----------------
#define BS   512
#define NN   128
#define HX   256
#define HL   64
#define HY   128
#define M_TOT (BS * NN)        // 65536 rows
#define KIN   (HX + HL)        // 320
#define KUP   (HX + HL + HY)   // 448
#define LN_EPS 1e-5f

typedef __bf16 bf16;
typedef __attribute__((ext_vector_type(16))) __bf16 v16bf;
typedef __attribute__((ext_vector_type(8)))  float  v8f;
typedef __attribute__((ext_vector_type(4)))  int    v4i;

union FragU { v16bf v; bf16 u[16]; };

// ---------------- optional CDNA5 async global->LDS copy ----------------
#if defined(__has_builtin)
# if __has_builtin(__builtin_amdgcn_global_load_async_to_lds_b128) && \
     __has_builtin(__builtin_amdgcn_s_wait_asynccnt)
#  define HAVE_ASYNC_LDS 1
# endif
#endif
#ifndef HAVE_ASYNC_LDS
# define HAVE_ASYNC_LDS 0
#endif

// copy 16 bytes (8 bf16) global -> LDS
__device__ __forceinline__ void copy16_to_lds(const bf16* g, bf16* l) {
#if HAVE_ASYNC_LDS
    __builtin_amdgcn_global_load_async_to_lds_b128(
        (__attribute__((address_space(1))) v4i*)(g),
        (__attribute__((address_space(3))) v4i*)(l), 0, 0);
#else
    *(uint4*)l = *(const uint4*)g;
#endif
}

__device__ __forceinline__ void wait_async_copies() {
#if HAVE_ASYNC_LDS
    __builtin_amdgcn_s_wait_asynccnt(0);
#endif
}

// A fragment (16x32 bf16, row M per lane, K split per lane-half; ISA 7.12.2)
__device__ __forceinline__ v16bf load_a_frag(const bf16* Ash, int ldk, int rowBase) {
    const int lane = threadIdx.x & 31;
    const int m  = lane & 15;
    const int k0 = (lane >> 4) << 3;   // lanes 0-15: K 0..7/16..23 ; 16-31: 8..15/24..31
    FragU f;
    const bf16* p = Ash + (rowBase + m) * ldk + k0;
#pragma unroll
    for (int t = 0; t < 8; ++t) { f.u[t] = p[t]; f.u[8 + t] = p[16 + t]; }
    return f.v;
}

// B fragment (32x16 bf16, column N per lane; BshT stored [n][k] so reads are contiguous)
__device__ __forceinline__ v16bf load_b_frag(const bf16* BshT, int ldk, int colBase) {
    const int lane = threadIdx.x & 31;
    const int n  = lane & 15;
    const int k0 = (lane >> 4) << 3;   // lanes 0-15: K 0..7/16..23 ; 16-31: 8..15/24..31
    FragU f;
    const bf16* p = BshT + (colBase + n) * ldk + k0;
#pragma unroll
    for (int t = 0; t < 8; ++t) { f.u[t] = p[t]; f.u[8 + t] = p[16 + t]; }
    return f.v;
}

#define WMMA_BF16(A, B, C) \
    __builtin_amdgcn_wmma_f32_16x16x32_bf16(false, (A), false, (B), (short)0, (C), false, false)

// ---------------- K1: degree^-1/2 per (b,i) ----------------
__global__ __launch_bounds__(256) void k_dinv(const int* __restrict__ E,
                                              const unsigned char* __restrict__ mask,
                                              float* __restrict__ dinv) {
    const int wave = threadIdx.x >> 5;
    const int lane = threadIdx.x & 31;
    const int row  = blockIdx.x * 8 + wave;        // b*NN + i
    const int b = row >> 7, i = row & 127;
    const float mi = mask[row] ? 1.f : 0.f;
    float s = 0.f;
#pragma unroll
    for (int t = 0; t < 4; ++t) {
        const int j = lane + t * 32;
        const float mj = mask[b * NN + j] ? 1.f : 0.f;
        s += (E[(((size_t)b * NN + i) * NN + j) * 2 + 1] != 0) ? (mi * mj) : 0.f;
    }
#pragma unroll
    for (int off = 16; off > 0; off >>= 1) s += __shfl_xor(s, off, 32);
    if (lane == 0) {
        const float deg = s + 1.f;                 // A = adj + I
        dinv[row] = (deg > 0.f) ? rsqrtf(fmaxf(deg, 1.f)) : 0.f;
    }
}

// ---------------- K2: C_bf16[M x N] = concat(A0,A1) @ W  (64x64 tile, 8 waves) -------------
__global__ __launch_bounds__(256) void k_gemm_in(const float* __restrict__ A0, int K0,
                                                 const float* __restrict__ A1, int Kc,
                                                 const float* __restrict__ W, int N,
                                                 bf16* __restrict__ Cout) {
    __shared__ bf16 Ash[64 * 32];
    __shared__ bf16 BshT[64 * 32];

    const int tid = threadIdx.x;
    const int rowTileBase = blockIdx.x * 64;
    const int colTileBase = blockIdx.y * 64;
    const int wave = tid >> 5;
    const int waveRow = (wave & 3) * 16;
    const int waveCol = (wave >> 2) * 32;
    const int K1s = Kc - K0;

    v8f acc0 = {}, acc1 = {};

    for (int kb = 0; kb < Kc; kb += 32) {
        { // A tile 64x32: gather from concat, fp32 -> bf16 (native cvt)
            const int rr = tid >> 2, kk0 = (tid & 3) * 8;
            const int row = rowTileBase + rr;
#pragma unroll
            for (int t = 0; t < 8; ++t) {
                const int kg = kb + kk0 + t;
                const float v = (kg < K0) ? A0[(size_t)row * K0 + kg]
                                          : A1[(size_t)row * K1s + (kg - K0)];
                Ash[rr * 32 + kk0 + t] = (bf16)v;
            }
        }
        { // B tile 32x64 -> transposed LDS [n][k]
            const int kk = tid >> 3, nn0 = (tid & 7) * 8;
#pragma unroll
            for (int t = 0; t < 8; ++t)
                BshT[(nn0 + t) * 32 + kk] = (bf16)W[(size_t)(kb + kk) * N + colTileBase + nn0 + t];
        }
        __syncthreads();
        const v16bf a  = load_a_frag(Ash, 32, waveRow);
        const v16bf b0 = load_b_frag(BshT, 32, waveCol);
        const v16bf b1 = load_b_frag(BshT, 32, waveCol + 16);
        acc0 = WMMA_BF16(a, b0, acc0);
        acc1 = WMMA_BF16(a, b1, acc1);
        __syncthreads();
    }

    const int lane = tid & 31, mOff = (lane >> 4) * 8, nIdx = lane & 15;
#pragma unroll
    for (int r = 0; r < 8; ++r) {
        const size_t row = rowTileBase + waveRow + r + mOff;
        const int c0 = colTileBase + waveCol + nIdx;
        Cout[row * N + c0]      = (bf16)acc0[r];
        Cout[row * N + c0 + 16] = (bf16)acc1[r];
    }
}

// ---------------- K3: [Xa | la] = An @ [XW | lW] + bias, An built on the fly -------------
__global__ __launch_bounds__(256) void k_aggr(const int* __restrict__ E,
                                              const unsigned char* __restrict__ mask,
                                              const float* __restrict__ dinv,
                                              const bf16* __restrict__ XW,
                                              const bf16* __restrict__ lW,
                                              const float* __restrict__ bx,
                                              const float* __restrict__ bl,
                                              bf16* __restrict__ Xa,
                                              bf16* __restrict__ la) {
    __shared__ bf16 Ash[64 * 32];
    __shared__ bf16 BshT[64 * 32];

    const int tid = threadIdx.x;
    const int b = blockIdx.x;
    const int rowTileBase = blockIdx.y * 64;       // 0 | 64
    const int colTileBase = blockIdx.z * 64;       // 0..256, concat cols of [XW(256) | lW(64)]
    const int wave = tid >> 5;
    const int waveRow = (wave & 3) * 16;
    const int waveCol = (wave >> 2) * 32;

    v8f acc0 = {}, acc1 = {};

    for (int kb = 0; kb < NN; kb += 32) {
        { // An tile 64x32: ((E!=0)*mi*mj + I) * di*dj  -> bf16
            const int rr = tid >> 2, kk0 = (tid & 3) * 8;
            const int i = rowTileBase + rr;
            const float di = dinv[b * NN + i];
            const float mi = mask[b * NN + i] ? 1.f : 0.f;
#pragma unroll
            for (int t = 0; t < 8; ++t) {
                const int j = kb + kk0 + t;
                const float mj  = mask[b * NN + j] ? 1.f : 0.f;
                const float adj = (E[(((size_t)b * NN + i) * NN + j) * 2 + 1] != 0) ? (mi * mj) : 0.f;
                const float Aij = adj + ((i == j) ? 1.f : 0.f);
                Ash[rr * 32 + kk0 + t] = (bf16)(Aij * di * dinv[b * NN + j]);
            }
        }
        { // B tile: rows j of XW/lW for this batch -> transposed LDS (pure bf16 moves)
            const int kk = tid >> 3, nn0 = (tid & 7) * 8;
            const size_t rowIdx = (size_t)b * NN + (kb + kk);
#pragma unroll
            for (int t = 0; t < 8; ++t) {
                const int c = colTileBase + nn0 + t;
                BshT[(nn0 + t) * 32 + kk] =
                    (c < HX) ? XW[rowIdx * HX + c] : lW[rowIdx * HL + (c - HX)];
            }
        }
        __syncthreads();
        const v16bf a  = load_a_frag(Ash, 32, waveRow);
        const v16bf b0 = load_b_frag(BshT, 32, waveCol);
        const v16bf b1 = load_b_frag(BshT, 32, waveCol + 16);
        acc0 = WMMA_BF16(a, b0, acc0);
        acc1 = WMMA_BF16(a, b1, acc1);
        __syncthreads();
    }

    const int lane = tid & 31, mOff = (lane >> 4) * 8, nIdx = lane & 15;
#pragma unroll
    for (int r = 0; r < 8; ++r) {
        const int i = rowTileBase + waveRow + r + mOff;
        const size_t row = (size_t)b * NN + i;
#pragma unroll
        for (int cc = 0; cc < 2; ++cc) {
            const int c = colTileBase + waveCol + cc * 16 + nIdx;
            const float v = (cc == 0) ? acc0[r] : acc1[r];
            if (c < HX) Xa[row * HX + c]        = (bf16)(v + bx[c]);
            else        la[row * HL + (c - HX)] = (bf16)(v + bl[c - HX]);
        }
    }
}

// ---------------- K4: X_out = LN(relu(concat(Xa,la,y) @ W1 + b1)) ----------------
__global__ __launch_bounds__(256) void k_update_X(const bf16* __restrict__ Xa,
                                                  const bf16* __restrict__ la,
                                                  const float* __restrict__ y,
                                                  const float* __restrict__ W1,
                                                  const float* __restrict__ b1,
                                                  const float* __restrict__ g1,
                                                  const float* __restrict__ be1,
                                                  float* __restrict__ out) {
    __shared__ bf16 Ash[16 * 32];
    __shared__ bf16 BshT[256 * 32];
    __shared__ float Csh[16 * 257];
    __shared__ float muS[16], rsS[16];

    const int tid = threadIdx.x;
    const int rowBase = blockIdx.x * 16;
    const int wave = tid >> 5, lane = tid & 31;
    const int waveCol = wave * 32;

    v8f acc0 = {}, acc1 = {};

    for (int kb = 0; kb < KUP; kb += 32) {
        // A tile 16x32 from concat [Xa(256) | la(64) | y(128, row-broadcast)]
        if (kb < HX + HL) {
            // straight bf16 copy: async global->LDS, 16B per issuing thread
            if (tid < 64) {
                const int rr = tid >> 2, seg = tid & 3;
                const int row = rowBase + rr;
                const bf16* src = (kb < HX)
                    ? Xa + (size_t)row * HX + kb + seg * 8
                    : la + (size_t)row * HL + (kb - HX) + seg * 8;
                copy16_to_lds(src, &Ash[rr * 32 + seg * 8]);
            }
        } else {
            // y segment: fp32 -> bf16 convert
            const int rr = tid >> 4, kk0 = (tid & 15) * 2;
            const int bb = (rowBase + rr) >> 7;
#pragma unroll
            for (int t = 0; t < 2; ++t) {
                const int kg = kb + kk0 + t;
                Ash[rr * 32 + kk0 + t] = (bf16)y[(size_t)bb * HY + (kg - HX - HL)];
            }
        }
        { // B tile 32x256 from W1, transposed; prefetch next tile
            const int kk = tid >> 3, nn0 = (tid & 7) * 32;
            if (kb + 32 < KUP)
                __builtin_prefetch(&W1[(size_t)(kb + 32 + kk) * HX + nn0], 0, 0);
#pragma unroll
            for (int t = 0; t < 32; ++t)
                BshT[(nn0 + t) * 32 + kk] = (bf16)W1[(size_t)(kb + kk) * HX + nn0 + t];
        }
        wait_async_copies();
        __syncthreads();
        const v16bf a  = load_a_frag(Ash, 32, 0);
        const v16bf b0 = load_b_frag(BshT, 32, waveCol);
        const v16bf b1 = load_b_frag(BshT, 32, waveCol + 16);
        acc0 = WMMA_BF16(a, b0, acc0);
        acc1 = WMMA_BF16(a, b1, acc1);
        __syncthreads();
    }

    // bias + ReLU into LDS row buffer
    const int mOff = (lane >> 4) * 8, nIdx = lane & 15;
#pragma unroll
    for (int r = 0; r < 8; ++r) {
        const int rw = r + mOff, c0 = waveCol + nIdx;
        Csh[rw * 257 + c0]      = fmaxf(acc0[r] + b1[c0], 0.f);
        Csh[rw * 257 + c0 + 16] = fmaxf(acc1[r] + b1[c0 + 16], 0.f);
    }
    __syncthreads();

    // LayerNorm stats: each wave owns 2 rows of 256
    for (int rr = wave * 2; rr < wave * 2 + 2; ++rr) {
        float s = 0.f, sq = 0.f;
#pragma unroll
        for (int t = 0; t < 8; ++t) {
            const float v = Csh[rr * 257 + lane * 8 + t];
            s += v; sq += v * v;
        }
#pragma unroll
        for (int off = 16; off > 0; off >>= 1) {
            s  += __shfl_xor(s, off, 32);
            sq += __shfl_xor(sq, off, 32);
        }
        if (lane == 0) {
            const float mu  = s * (1.f / HX);
            const float var = sq * (1.f / HX) - mu * mu;
            muS[rr] = mu; rsS[rr] = rsqrtf(var + LN_EPS);
        }
    }
    __syncthreads();

    const int rw = tid >> 4, c0 = (tid & 15) * 16;
    const float mu = muS[rw], rs = rsS[rw];
#pragma unroll
    for (int t = 0; t < 16; ++t) {
        const int c = c0 + t;
        out[(size_t)(rowBase + rw) * HX + c] = (Csh[rw * 257 + c] - mu) * rs * g1[c] + be1[c];
    }
}

// ---------------- K5: label_out = LN(relu(la @ W2 + b2)) ----------------
__global__ __launch_bounds__(128) void k_update_label(const bf16* __restrict__ la,
                                                      const float* __restrict__ W2,
                                                      const float* __restrict__ b2,
                                                      const float* __restrict__ g2,
                                                      const float* __restrict__ be2,
                                                      float* __restrict__ out) {
    __shared__ bf16 Ash[16 * 32];
    __shared__ bf16 BshT[64 * 32];
    __shared__ float Csh[16 * 65];
    __shared__ float muS[16], rsS[16];

    const int tid = threadIdx.x;
    const int rowBase = blockIdx.x * 16;
    const int wave = tid >> 5, lane = tid & 31;
    const int waveCol = wave * 16;

    v8f acc = {};

    for (int kb = 0; kb < HL; kb += 32) {
        // A tile 16x32: straight bf16 copy via async global->LDS
        if (tid < 64) {
            const int rr = tid >> 2, seg = tid & 3;
            copy16_to_lds(la + (size_t)(rowBase + rr) * HL + kb + seg * 8,
                          &Ash[rr * 32 + seg * 8]);
        }
        { // B tile 32x64, transposed, fp32 -> bf16
            const int kk = tid >> 2, nn0 = (tid & 3) * 16;
#pragma unroll
            for (int t = 0; t < 16; ++t)
                BshT[(nn0 + t) * 32 + kk] = (bf16)W2[(size_t)(kb + kk) * HL + nn0 + t];
        }
        wait_async_copies();
        __syncthreads();
        const v16bf a = load_a_frag(Ash, 32, 0);
        const v16bf b = load_b_frag(BshT, 32, waveCol);
        acc = WMMA_BF16(a, b, acc);
        __syncthreads();
    }

    const int mOff = (lane >> 4) * 8, nIdx = lane & 15;
#pragma unroll
    for (int r = 0; r < 8; ++r) {
        const int rw = r + mOff, c = waveCol + nIdx;
        Csh[rw * 65 + c] = fmaxf(acc[r] + b2[c], 0.f);
    }
    __syncthreads();

    for (int rr = wave * 4; rr < wave * 4 + 4; ++rr) {
        float s = 0.f, sq = 0.f;
#pragma unroll
        for (int t = 0; t < 2; ++t) {
            const float v = Csh[rr * 65 + lane * 2 + t];
            s += v; sq += v * v;
        }
#pragma unroll
        for (int off = 16; off > 0; off >>= 1) {
            s  += __shfl_xor(s, off, 32);
            sq += __shfl_xor(sq, off, 32);
        }
        if (lane == 0) {
            const float mu  = s * (1.f / HL);
            const float var = sq * (1.f / HL) - mu * mu;
            muS[rr] = mu; rsS[rr] = rsqrtf(var + LN_EPS);
        }
    }
    __syncthreads();

    const int rw = tid >> 3, c0 = (tid & 7) * 8;
    const float mu = muS[rw], rs = rsS[rw];
#pragma unroll
    for (int t = 0; t < 8; ++t) {
        const int c = c0 + t;
        out[(size_t)(rowBase + rw) * HL + c] = (Csh[rw * 65 + c] - mu) * rs * g2[c] + be2[c];
    }
}

// ---------------- host launcher ----------------
extern "C" void kernel_launch(void* const* d_in, const int* in_sizes, int n_in,
                              void* d_out, int out_size, void* d_ws, size_t ws_size,
                              hipStream_t stream) {
    (void)in_sizes; (void)n_in; (void)out_size; (void)ws_size;

    const float* X      = (const float*)d_in[0];
    const int*   E      = (const int*)d_in[1];
    const float* y      = (const float*)d_in[2];
    const float* label  = (const float*)d_in[3];
    const unsigned char* mask = (const unsigned char*)d_in[4];
    const float* Wx  = (const float*)d_in[5];
    const float* bx  = (const float*)d_in[6];
    const float* Wl  = (const float*)d_in[7];
    const float* bl  = (const float*)d_in[8];
    const float* W1  = (const float*)d_in[9];
    const float* b1  = (const float*)d_in[10];
    const float* g1  = (const float*)d_in[11];
    const float* be1 = (const float*)d_in[12];
    const float* W2  = (const float*)d_in[13];
    const float* b2  = (const float*)d_in[14];
    const float* g2  = (const float*)d_in[15];
    const float* be2 = (const float*)d_in[16];

    // workspace layout (~84 MB): dinv f32 | XW bf16 | lW bf16 | Xa bf16 | la bf16
    char* ws = (char*)d_ws;
    float* dinv = (float*)ws; ws += (size_t)M_TOT * 4;
    bf16* XW    = (bf16*)ws;  ws += (size_t)M_TOT * HX * 2;
    bf16* lW    = (bf16*)ws;  ws += (size_t)M_TOT * HL * 2;
    bf16* Xa    = (bf16*)ws;  ws += (size_t)M_TOT * HX * 2;
    bf16* laB   = (bf16*)ws;  ws += (size_t)M_TOT * HL * 2;

    k_dinv<<<M_TOT / 8, 256, 0, stream>>>(E, mask, dinv);
    k_gemm_in<<<dim3(M_TOT / 64, HX / 64), 256, 0, stream>>>(X, HX, label, KIN, Wx, HX, XW);
    k_gemm_in<<<dim3(M_TOT / 64, HL / 64), 256, 0, stream>>>(label, HL, label, HL, Wl, HL, lW);
    k_aggr<<<dim3(BS, NN / 64, KIN / 64), 256, 0, stream>>>(E, mask, dinv, XW, lW, bx, bl, Xa, laB);
    k_update_X<<<M_TOT / 16, 256, 0, stream>>>(Xa, laB, y, W1, b1, g1, be1, (float*)d_out);
    k_update_label<<<M_TOT / 16, 128, 0, stream>>>(laB, W2, b2, g2, be2,
                                                   (float*)d_out + (size_t)M_TOT * HX);
}
